// AttentionBlock_76063870812721
// MI455X (gfx1250) — matrix operand order repaired and checked by reading, not compiled, not run on hardware
//
#include <hip/hip_runtime.h>
#include <hip/hip_bf16.h>

// ---------------------------------------------------------------------------
// Types for CDNA5 WMMA (wave32, 16x16x32 bf16 -> f32)
// ---------------------------------------------------------------------------
typedef __attribute__((ext_vector_type(16))) __bf16 v16bf;
typedef __attribute__((ext_vector_type(8)))  float  v8f;
typedef __attribute__((ext_vector_type(4)))  unsigned int u32x4;
typedef int async_vec __attribute__((vector_size(16)));   // matches builtin param

#define AS1 __attribute__((address_space(1)))
#define AS3 __attribute__((address_space(3)))

#if defined(__has_builtin)
#if __has_builtin(__builtin_amdgcn_global_load_async_to_lds_b128)
#define HAVE_ASYNC_LDS 1
#endif
#endif
#ifndef HAVE_ASYNC_LDS
#define HAVE_ASYNC_LDS 0
#endif

union FragAB { u32x4 q[2]; v16bf v; };
union Pack8  { unsigned short h[8]; u32x4 q; };

__device__ __forceinline__ unsigned short f2bf(float f) {
  unsigned int u = __float_as_uint(f);
  u += 0x7FFFu + ((u >> 16) & 1u);           // round-to-nearest-even
  return (unsigned short)(u >> 16);
}

__device__ __forceinline__ v8f wmma_bf16(v16bf a, v16bf b, v8f c) {
  return __builtin_amdgcn_wmma_f32_16x16x32_bf16(false, a, false, b,
                                                 (short)0, c, false, false);
}

// A-fragment 16x32 bf16 (rows contiguous in K).
// Per-lane: m = lane&15, kb = (lane<16)?0:16; 16 contiguous bf16 at A[m][kb..].
__device__ __forceinline__ v16bf load_a_frag(const unsigned short* A, int ldk, int lane) {
  int m  = lane & 15;
  int kb = (lane < 16) ? 0 : 16;
  const unsigned short* p = A + (size_t)m * ldk + kb;
  FragAB f;
  f.q[0] = *(const u32x4*)(p);
  f.q[1] = *(const u32x4*)(p + 8);
  return f.v;
}

// B-fragment 32x16 bf16 from B^T (rows = N, contiguous in K).
// Per-lane: n = lane&15, ks = (lane<16)?0:16; 16 contiguous bf16 at Bt[n][ks..].
__device__ __forceinline__ v16bf load_b_frag(const unsigned short* Bt, int ldk, int lane) {
  int n  = lane & 15;
  int ks = (lane < 16) ? 0 : 16;
  const unsigned short* p = Bt + (size_t)n * ldk + ks;
  FragAB f;
  f.q[0] = *(const u32x4*)(p);
  f.q[1] = *(const u32x4*)(p + 8);
  return f.v;
}

// ---------------------------------------------------------------------------
// Problem constants
// ---------------------------------------------------------------------------
#define BATCH 4
#define C     256
#define NPIX  4096        // 64*64
#define HEADS 4
#define DH    64
#define GROUPS 8
#define CG    32          // channels per group
#define ATT_SCALE 0.03125f  // (C*HEADS)^-0.5 = 1/32

// ---------------------------------------------------------------------------
// K0: fp32 -> bf16 weight conversion
// ---------------------------------------------------------------------------
__global__ void cvt_weights(const float* __restrict__ wqkv,
                            const float* __restrict__ wproj,
                            unsigned short* __restrict__ wqkv_bf,
                            unsigned short* __restrict__ wproj_bf) {
  int i = blockIdx.x * blockDim.x + threadIdx.x;
  if (i < 3 * C * C) wqkv_bf[i] = f2bf(wqkv[i]);
  if (i < C * C)     wproj_bf[i] = f2bf(wproj[i]);
}

// ---------------------------------------------------------------------------
// K1: GroupNorm -> xn^T [b][p][c] bf16 (B^T layout for the QKV GEMM)
// ---------------------------------------------------------------------------
__global__ void groupnorm_kernel(const float* __restrict__ x,
                                 const float* __restrict__ gamma,
                                 const float* __restrict__ beta,
                                 unsigned short* __restrict__ xnT) {
  int b = blockIdx.x >> 3;
  int g = blockIdx.x & 7;
  int tid = threadIdx.x;
  const float* xg = x + ((size_t)b * C + g * CG) * NPIX;

  float s = 0.f, ss = 0.f;
  for (int i = tid; i < CG * NPIX; i += 256) {
    float v = xg[i];
    s += v; ss += v * v;
  }
  __shared__ float red[256], red2[256];
  red[tid] = s; red2[tid] = ss;
  __syncthreads();
  for (int o = 128; o > 0; o >>= 1) {
    if (tid < o) { red[tid] += red[tid + o]; red2[tid] += red2[tid + o]; }
    __syncthreads();
  }
  const float inv = 1.f / (float)(CG * NPIX);
  float mean = red[0] * inv;
  float var  = red2[0] * inv - mean * mean;
  float rstd = rsqrtf(var + 1e-5f);

  float gam[CG], bet[CG];
#pragma unroll
  for (int cl = 0; cl < CG; ++cl) {
    gam[cl] = gamma[g * CG + cl] * rstd;
    bet[cl] = beta[g * CG + cl];
  }
  for (int p = tid; p < NPIX; p += 256) {
    Pack8 pk[4];
#pragma unroll
    for (int cl = 0; cl < CG; ++cl) {
      float y = (xg[(size_t)cl * NPIX + p] - mean) * gam[cl] + bet[cl];
      pk[cl >> 3].h[cl & 7] = f2bf(y);
    }
    u32x4* dst = (u32x4*)(xnT + ((size_t)b * NPIX + p) * C + g * CG);
    dst[0] = pk[0].q; dst[1] = pk[1].q; dst[2] = pk[2].q; dst[3] = pk[3].q;
  }
}

// ---------------------------------------------------------------------------
// K2: QKV GEMM (one wave -> 16x64 tile), epilogue scatters to WMMA layouts:
//   q,k -> [b][h][n][dh]   v -> [b][h][dh][n]
// ---------------------------------------------------------------------------
__global__ void qkv_gemm(const unsigned short* __restrict__ W,
                         const unsigned short* __restrict__ Xt,
                         unsigned short* __restrict__ qT,
                         unsigned short* __restrict__ kT,
                         unsigned short* __restrict__ vM) {
  int wid  = (blockIdx.x * blockDim.x + threadIdx.x) >> 5;
  int lane = threadIdx.x & 31;
  int b  = wid / (48 * 64);
  int r  = wid % (48 * 64);
  int m0 = (r / 64) * 16;
  int n0 = (r % 64) * 64;

  const unsigned short* Arow = W + (size_t)m0 * C;
  const unsigned short* Bt   = Xt + (size_t)b * NPIX * C;

  v8f zero = {0.f,0.f,0.f,0.f,0.f,0.f,0.f,0.f};
  v8f acc[4] = {zero, zero, zero, zero};

  for (int k0 = 0; k0 < C; k0 += 32) {
    if (k0 + 32 < C) __builtin_prefetch(Arow + k0 + 32, 0, 3);  // WGP-scope prefetch
    v16bf a = load_a_frag(Arow + k0, C, lane);
#pragma unroll
    for (int t = 0; t < 4; ++t) {
      v16bf bf = load_b_frag(Bt + (size_t)(n0 + 16 * t) * C + k0, C, lane);
      acc[t] = wmma_bf16(a, bf, acc[t]);
    }
  }

  int half8 = (lane < 16) ? 0 : 8;
#pragma unroll
  for (int t = 0; t < 4; ++t) {
    int p = n0 + t * 16 + (lane & 15);
    if (m0 < 2 * C) {                     // q or k: packed b128 store
      int mo = (m0 < C) ? m0 : (m0 - C);
      unsigned short* base = (m0 < C) ? qT : kT;
      int head = mo >> 6;
      int c0   = (mo & 63) + half8;
      Pack8 pk;
#pragma unroll
      for (int rr = 0; rr < 8; ++rr) pk.h[rr] = f2bf(acc[t][rr]);
      *(u32x4*)(base + (((size_t)b * HEADS + head) * NPIX + p) * DH + c0) = pk.q;
    } else {                              // v: strided store
      int mo = m0 - 2 * C;
      int head = mo >> 6;
#pragma unroll
      for (int rr = 0; rr < 8; ++rr) {
        int cc = (mo & 63) + half8 + rr;
        vM[(((size_t)b * HEADS + head) * DH + cc) * NPIX + p] = f2bf(acc[t][rr]);
      }
    }
  }
}

// ---------------------------------------------------------------------------
// K3: cooperative flash attention.
// Block = 4 waves; per 64-key step K/V tiles are staged into double-buffered
// LDS (async global->LDS), shared by all 4 query tiles.
// ---------------------------------------------------------------------------
__device__ __forceinline__ void stage_kv(const unsigned short* __restrict__ Kg,
                                         const unsigned short* __restrict__ Vg,
                                         unsigned short* Kl, unsigned short* Vl,
                                         int tid) {
#if HAVE_ASYNC_LDS
#pragma unroll
  for (int t = 0; t < 4; ++t) {             // K tile: 8KB contiguous
    int idx = tid + t * 128;
    __builtin_amdgcn_global_load_async_to_lds_b128(
        (AS1 async_vec*)(Kg + idx * 8),
        (AS3 async_vec*)(Kl + idx * 8), 0, 0);
  }
#pragma unroll
  for (int t = 0; t < 4; ++t) {             // V tile: 64 rows of 128B
    int idx = tid + t * 128;
    int row = idx >> 3;
    int col = (idx & 7) * 8;
    __builtin_amdgcn_global_load_async_to_lds_b128(
        (AS1 async_vec*)(Vg + (size_t)row * NPIX + col),
        (AS3 async_vec*)(Vl + row * 64 + col), 0, 0);
  }
#else
#pragma unroll
  for (int t = 0; t < 4; ++t) {
    int idx = tid + t * 128;
    *(u32x4*)(Kl + idx * 8) = *(const u32x4*)(Kg + idx * 8);
  }
#pragma unroll
  for (int t = 0; t < 4; ++t) {
    int idx = tid + t * 128;
    int row = idx >> 3;
    int col = (idx & 7) * 8;
    *(u32x4*)(Vl + row * 64 + col) = *(const u32x4*)(Vg + (size_t)row * NPIX + col);
  }
#endif
}

__device__ __forceinline__ void wait_async() {
#if HAVE_ASYNC_LDS
  asm volatile("s_wait_asynccnt 0x0" ::: "memory");
#endif
}

__global__ void __launch_bounds__(128) attn_kernel(
    const unsigned short* __restrict__ qT,
    const unsigned short* __restrict__ kT,
    const unsigned short* __restrict__ vM,
    unsigned short* __restrict__ aoT) {
  int blk = blockIdx.x;            // [b(2b) | h(2b) | itg(6b)]
  int itg = blk & 63;
  int h   = (blk >> 6) & 3;
  int b   = blk >> 8;
  int tid   = threadIdx.x;
  int wave  = tid >> 5;
  int lane  = tid & 31;
  int ln15  = lane & 15;
  int half8 = (lane < 16) ? 0 : 8;
  int ks    = (lane < 16) ? 0 : 16;

  __shared__ __align__(16) unsigned short Klds[2][64 * 64];
  __shared__ __align__(16) unsigned short Vlds[2][64 * 64];
  __shared__ __align__(16) unsigned short Pb[4][16 * 64];
  __shared__ float aBuf[4][16];
  __shared__ float lBuf[4][16];

  size_t bh = (size_t)b * HEADS + h;
  int it = itg * 4 + wave;
  const unsigned short* Aq = qT + (bh * NPIX + (size_t)it * 16) * DH;
  const unsigned short* Kg = kT + bh * NPIX * DH;   // [j][64]
  const unsigned short* Vg = vM + bh * DH * NPIX;   // [c][4096]

  v16bf aq0 = load_a_frag(Aq + 0,  DH, lane);       // c = 0..31
  v16bf aq1 = load_a_frag(Aq + 32, DH, lane);       // c = 32..63

  float mstate[8], lstate[8];
#pragma unroll
  for (int r = 0; r < 8; ++r) { mstate[r] = -1e30f; lstate[r] = 0.f; }
  v8f zero = {0.f,0.f,0.f,0.f,0.f,0.f,0.f,0.f};
  v8f oacc[4] = {zero, zero, zero, zero};

  stage_kv(Kg, Vg, Klds[0], Vlds[0], tid);
  wait_async();
  __syncthreads();

  const int NSTEP = NPIX / 64;
  for (int s = 0; s < NSTEP; ++s) {
    int cur = s & 1;
    if (s + 1 < NSTEP)
      stage_kv(Kg + (size_t)(s + 1) * 64 * DH, Vg + (size_t)(s + 1) * 64,
               Klds[cur ^ 1], Vlds[cur ^ 1], tid);

    // ---- scores: four 16-wide key tiles from the shared K tile ----
    v8f sarr[4];
#pragma unroll
    for (int js = 0; js < 4; ++js) {
      const unsigned short* krow = &Klds[cur][(js * 16) * 64];
      v16bf b0 = load_b_frag(krow + 0,  64, lane);
      v16bf b1 = load_b_frag(krow + 32, 64, lane);
      sarr[js] = wmma_bf16(aq1, b1, wmma_bf16(aq0, b0, zero));
    }

    // ---- online softmax over 64 keys ----
    float al[8];
#pragma unroll
    for (int r = 0; r < 8; ++r) {
      float v0 = sarr[0][r] * ATT_SCALE;
      float v1 = sarr[1][r] * ATT_SCALE;
      float v2 = sarr[2][r] * ATT_SCALE;
      float v3 = sarr[3][r] * ATT_SCALE;
      float rm = fmaxf(fmaxf(v0, v1), fmaxf(v2, v3));
      rm = fmaxf(rm, __shfl_xor(rm, 1, 32));
      rm = fmaxf(rm, __shfl_xor(rm, 2, 32));
      rm = fmaxf(rm, __shfl_xor(rm, 4, 32));
      rm = fmaxf(rm, __shfl_xor(rm, 8, 32));
      float mn = fmaxf(mstate[r], rm);
      al[r] = __expf(mstate[r] - mn);
      float p0 = __expf(v0 - mn);
      float p1 = __expf(v1 - mn);
      float p2 = __expf(v2 - mn);
      float p3 = __expf(v3 - mn);
      float rs = (p0 + p1) + (p2 + p3);
      rs += __shfl_xor(rs, 1, 32);
      rs += __shfl_xor(rs, 2, 32);
      rs += __shfl_xor(rs, 4, 32);
      rs += __shfl_xor(rs, 8, 32);
      lstate[r] = lstate[r] * al[r] + rs;
      mstate[r] = mn;
      unsigned short* pr = &Pb[wave][(r + half8) * 64];   // transpose P -> [i][j]
      pr[ 0 + ln15] = f2bf(p0);
      pr[16 + ln15] = f2bf(p1);
      pr[32 + ln15] = f2bf(p2);
      pr[48 + ln15] = f2bf(p3);
    }
    if (lane == 0) {
#pragma unroll
      for (int r = 0; r < 8; ++r) aBuf[wave][r] = al[r];
    }
    if (lane == 16) {
#pragma unroll
      for (int r = 0; r < 8; ++r) aBuf[wave][8 + r] = al[r];
    }
    __builtin_amdgcn_wave_barrier();
    asm volatile("s_wait_dscnt 0x0" ::: "memory");

    // ---- O^T += V * P^T with online rescale ----
    float alpha = aBuf[wave][ln15];
#pragma unroll
    for (int mc = 0; mc < 4; ++mc) {
#pragma unroll
      for (int r = 0; r < 8; ++r) oacc[mc][r] *= alpha;
    }
#pragma unroll
    for (int jc = 0; jc < 2; ++jc) {
      FragAB fp;
      const u32x4* psrc = (const u32x4*)(&Pb[wave][ln15 * 64 + jc * 32 + ks]);
      fp.q[0] = psrc[0];
      fp.q[1] = psrc[1];
      v16bf bp = fp.v;
#pragma unroll
      for (int mc = 0; mc < 4; ++mc) {
        v16bf av = load_a_frag(&Vlds[cur][(mc * 16) * 64] + jc * 32, 64, lane);
        oacc[mc] = wmma_bf16(av, bp, oacc[mc]);
      }
    }
    wait_async();
    __syncthreads();
  }

  // ---- finalize: divide by l, store O^T as [p][c] bf16 ----
  if (lane == 0) {
#pragma unroll
    for (int r = 0; r < 8; ++r) lBuf[wave][r] = lstate[r];
  }
  if (lane == 16) {
#pragma unroll
    for (int r = 0; r < 8; ++r) lBuf[wave][8 + r] = lstate[r];
  }
  __builtin_amdgcn_wave_barrier();
  asm volatile("s_wait_dscnt 0x0" ::: "memory");
  float linv = 1.f / lBuf[wave][ln15];

  int p = it * 16 + ln15;
  unsigned short* dst = aoT + ((size_t)b * NPIX + p) * C + h * DH;
#pragma unroll
  for (int mc = 0; mc < 4; ++mc) {
    Pack8 pk;
#pragma unroll
    for (int r = 0; r < 8; ++r) pk.h[r] = f2bf(oacc[mc][r] * linv);
    *(u32x4*)(dst + mc * 16 + half8) = pk.q;
  }
}

// ---------------------------------------------------------------------------
// K4: projection GEMM + bias + residual (fp32 output).
// ---------------------------------------------------------------------------
__global__ void proj_gemm(const unsigned short* __restrict__ W,
                          const unsigned short* __restrict__ At,
                          const float* __restrict__ bias,
                          const float* __restrict__ x,
                          float* __restrict__ out) {
  int wid  = (blockIdx.x * blockDim.x + threadIdx.x) >> 5;
  int lane = threadIdx.x & 31;
  int b  = wid / (16 * 64);
  int r  = wid % (16 * 64);
  int m0 = (r / 64) * 16;
  int n0 = (r % 64) * 64;

  const unsigned short* Arow = W + (size_t)m0 * C;
  const unsigned short* Bt   = At + (size_t)b * NPIX * C;

  v8f zero = {0.f,0.f,0.f,0.f,0.f,0.f,0.f,0.f};
  v8f acc[4] = {zero, zero, zero, zero};

  for (int k0 = 0; k0 < C; k0 += 32) {
    if (k0 + 32 < C) __builtin_prefetch(Arow + k0 + 32, 0, 3);  // WGP-scope prefetch
    v16bf a = load_a_frag(Arow + k0, C, lane);
#pragma unroll
    for (int t = 0; t < 4; ++t) {
      v16bf bf = load_b_frag(Bt + (size_t)(n0 + 16 * t) * C + k0, C, lane);
      acc[t] = wmma_bf16(a, bf, acc[t]);
    }
  }

  int half8 = (lane < 16) ? 0 : 8;
#pragma unroll
  for (int t = 0; t < 4; ++t) {
    int p = n0 + t * 16 + (lane & 15);
#pragma unroll
    for (int rr = 0; rr < 8; ++rr) {
      int o = m0 + half8 + rr;
      size_t idx = ((size_t)b * C + o) * NPIX + p;
      out[idx] = acc[t][rr] + bias[o] + x[idx];
    }
  }
}

// ---------------------------------------------------------------------------
// Launch
// ---------------------------------------------------------------------------
extern "C" void kernel_launch(void* const* d_in, const int* in_sizes, int n_in,
                              void* d_out, int out_size, void* d_ws, size_t ws_size,
                              hipStream_t stream) {
  (void)in_sizes; (void)n_in; (void)out_size; (void)ws_size;
  const float* x     = (const float*)d_in[0];
  const float* gamma = (const float*)d_in[1];
  const float* beta  = (const float*)d_in[2];
  const float* wqkv  = (const float*)d_in[3];
  const float* wproj = (const float*)d_in[4];
  const float* bproj = (const float*)d_in[5];
  float* out = (float*)d_out;

  char* ws = (char*)d_ws;
  size_t off = 0;
  auto alloc = [&](size_t bytes) -> void* {
    void* p = ws + off;
    off += (bytes + 255) & ~(size_t)255;
    return p;
  };
  unsigned short* wqkv_bf  = (unsigned short*)alloc((size_t)3 * C * C * 2);
  unsigned short* wproj_bf = (unsigned short*)alloc((size_t)C * C * 2);
  unsigned short* xnT      = (unsigned short*)alloc((size_t)BATCH * NPIX * C * 2);
  unsigned short* qT       = (unsigned short*)alloc((size_t)BATCH * HEADS * NPIX * DH * 2);
  unsigned short* kT       = (unsigned short*)alloc((size_t)BATCH * HEADS * NPIX * DH * 2);
  unsigned short* vM       = (unsigned short*)alloc((size_t)BATCH * HEADS * DH * NPIX * 2);
  unsigned short* aoT      = (unsigned short*)alloc((size_t)BATCH * NPIX * C * 2);

  cvt_weights<<<(3 * C * C + 255) / 256, 256, 0, stream>>>(wqkv, wproj, wqkv_bf, wproj_bf);
  groupnorm_kernel<<<BATCH * GROUPS, 256, 0, stream>>>(x, gamma, beta, xnT);
  qkv_gemm<<<1536, 256, 0, stream>>>(wqkv_bf, xnT, qT, kT, vM);
  // Attention: 4 waves per block sharing K/V LDS tiles; 1024 blocks.
  attn_kernel<<<BATCH * HEADS * (NPIX / 64), 128, 0, stream>>>(qT, kT, vM, aoT);
  proj_gemm<<<512, 256, 0, stream>>>(wproj_bf, aoT, bproj, x, out);
}
